// LinearAttnFFN_75204877353640
// MI455X (gfx1250) — matrix operand order repaired and checked by reading, not compiled, not run on hardware
//
#include <hip/hip_runtime.h>
#include <hip/hip_bf16.h>

// ---------------------------------------------------------------------------
// LinearAttn + FFN block for MI455X (gfx1250, wave32, WMMA, async-to-LDS).
// Shapes fixed by the reference: B=16, C=512, P=4, N=1024, F=1024, NPOS=4096.
//
// GEMM operands live in global memory pre-swizzled into WMMA fragment order:
// main loop = double-buffered global_load_async_to_lds_b128 staging overlapped
// with ds_load_b128 fragment reads + v_wmma_f32_16x16x32_bf16.
// ---------------------------------------------------------------------------

typedef __attribute__((ext_vector_type(16))) __bf16 v16bf;
typedef __attribute__((ext_vector_type(8)))  __bf16 v8bf;
typedef __attribute__((ext_vector_type(8)))  float  v8f;

#define NB    16          // batch
#define NC    512         // channels
#define NP    4           // P
#define NN    1024        // N (softmax axis)
#define NF    1024        // FFN hidden
#define NPOS  4096        // P*N
#define PER_BATCH 2097152 // C*NPOS

static __device__ __forceinline__ __bf16 f2bf(float f) { return (__bf16)f; }
static __device__ __forceinline__ float  bf2f(__bf16 h) { return (float)h; }

static __device__ __forceinline__ void async_copy16(unsigned ldsOff, const void* g) {
  asm volatile("global_load_async_to_lds_b128 %0, %1, off"
               :: "v"(ldsOff), "v"(g) : "memory");
}

// Swizzled activation index: (32-K x 64-N) tiles, fragment order inside a tile
// with k = 16*(lane/16) + j (contiguous k, same order as wpack_idx).  One tile = 2048 bf16 = 4KB,
// contiguous in memory.  Tile order: kb-major, nb-minor (NPOS/64 = 64 tiles).
static __device__ __forceinline__ size_t act_idx(int c, int pos) {
  const int kb   = c >> 5;
  const int nb   = pos >> 6;
  const int lane = (pos & 15) + (((c >> 4) & 1) << 4);
  const int nt   = (pos >> 4) & 3;
  const int j    = c & 15;
  return ((size_t)(kb * 64 + nb) << 11) + (nt << 9) + (lane << 4) + j;
}

// Swizzled weight index: (16-M x 32-K) tiles in A-fragment order (512 bf16/tile).
static __device__ __forceinline__ size_t wpack_idx(int m, int k, int kTiles) {
  const int lane = (m & 15) + (((k >> 4) & 1) << 4);
  const int j    = k & 15;
  return ((size_t)((m >> 4) * kTiles + (k >> 5)) << 9) + (lane << 4) + j;
}

// ---------------------------------------------------------------------------
// One-time weight repack: f32 row-major (M x K) -> bf16 A-fragment order.
// ---------------------------------------------------------------------------
__global__ __launch_bounds__(256)
void wpack_k(const float* __restrict__ W, __bf16* __restrict__ out,
             int kShift /* log2(K) */) {
  const int i = blockIdx.x * 256 + threadIdx.x;
  const int K = 1 << kShift;
  const int m = i >> kShift, k = i & (K - 1);
  out[wpack_idx(m, k, K >> 5)] = f2bf(W[i]);
}

// ---------------------------------------------------------------------------
// GroupNorm statistics: two-stage deterministic reduction.
// ---------------------------------------------------------------------------
__global__ __launch_bounds__(256)
void gn_partial_k(const float* __restrict__ in, float* __restrict__ partials) {
  __shared__ float s1[256], s2[256];
  const int b = blockIdx.x >> 6, seg = blockIdx.x & 63, t = threadIdx.x;
  const size_t base = (size_t)b * PER_BATCH + (size_t)seg * 32768;
  float s = 0.f, q = 0.f;
  for (int i = t; i < 32768; i += 256) {
    float v = in[base + i];
    s += v; q += v * v;
  }
  s1[t] = s; s2[t] = q;
  __syncthreads();
  for (int off = 128; off; off >>= 1) {
    if (t < off) { s1[t] += s1[t + off]; s2[t] += s2[t + off]; }
    __syncthreads();
  }
  if (t == 0) {
    partials[blockIdx.x * 2 + 0] = s1[0];
    partials[blockIdx.x * 2 + 1] = s2[0];
  }
}

__global__ __launch_bounds__(64)
void gn_finalize_k(const float* __restrict__ partials, float* __restrict__ stats) {
  __shared__ float s1[64], s2[64];
  const int b = blockIdx.x, t = threadIdx.x;
  s1[t] = partials[(b * 64 + t) * 2 + 0];
  s2[t] = partials[(b * 64 + t) * 2 + 1];
  __syncthreads();
  for (int off = 32; off; off >>= 1) {
    if (t < off) { s1[t] += s1[t + off]; s2[t] += s2[t + off]; }
    __syncthreads();
  }
  if (t == 0) {
    const float inv = 1.f / (float)PER_BATCH;
    const float mu  = s1[0] * inv;
    const float var = s2[0] * inv - mu * mu;
    stats[b * 2 + 0] = mu;
    stats[b * 2 + 1] = rsqrtf(var + 1e-5f);
  }
}

// Normalize + affine -> swizzled bf16; 8 channels per lane -> one b128 store.
__global__ __launch_bounds__(256)
void norm_pack_k(const float* __restrict__ in, const float* __restrict__ stats,
                 const float* __restrict__ gamma, const float* __restrict__ beta,
                 __bf16* __restrict__ out) {
  const size_t i = (size_t)blockIdx.x * 256 + threadIdx.x;
  const int pos = (int)(i & 4095);
  const int c0  = (int)((i >> 12) & 63) << 3;
  const int b   = (int)(i >> 18);
  const float mu = stats[b * 2 + 0], rs = stats[b * 2 + 1];
  const float* src = in + (size_t)b * PER_BATCH + (size_t)c0 * NPOS + pos;
  v8bf o;
#pragma unroll
  for (int r = 0; r < 8; ++r) {
    const float g = gamma[c0 + r] * rs;
    o[r] = f2bf(src[(size_t)r * NPOS] * g + (beta[c0 + r] - mu * g));
  }
  *(v8bf*)(out + (size_t)b * PER_BATCH + act_idx(c0, pos)) = o;
}

// q[b,pn] = w_q . y1[b,:,pn] + b_q   (row 0 of qkv_w), swizzled reads.
__global__ __launch_bounds__(256)
void q_k(const __bf16* __restrict__ y, const float* __restrict__ qkv_w,
         const float* __restrict__ qkv_b, float* __restrict__ q) {
  __shared__ float wq[NC];
  const int t = threadIdx.x;
  wq[t] = qkv_w[t]; wq[t + 256] = qkv_w[t + 256];
  __syncthreads();
  const int b = blockIdx.y;
  const int pn = blockIdx.x * 256 + t;
  const __bf16* yb = y + (size_t)b * PER_BATCH;
  float acc = qkv_b[0];
  for (int c = 0; c < NC; ++c) acc += wq[c] * bf2f(yb[act_idx(c, pn)]);
  q[(size_t)b * NPOS + pn] = acc;
}

// softmax over N=1024 per (b,p) row, in place.
__global__ __launch_bounds__(256)
void softmax_k(float* __restrict__ q) {
  __shared__ float red[256];
  const int t = threadIdx.x;
  float* row = q + (size_t)blockIdx.x * NN;
  float v[4];
  float m = -3.4e38f;
#pragma unroll
  for (int j = 0; j < 4; ++j) { v[j] = row[t + j * 256]; m = fmaxf(m, v[j]); }
  red[t] = m; __syncthreads();
  for (int off = 128; off; off >>= 1) {
    if (t < off) red[t] = fmaxf(red[t], red[t + off]);
    __syncthreads();
  }
  m = red[0]; __syncthreads();
  float s = 0.f;
#pragma unroll
  for (int j = 0; j < 4; ++j) { v[j] = __expf(v[j] - m); s += v[j]; }
  red[t] = s; __syncthreads();
  for (int off = 128; off; off >>= 1) {
    if (t < off) red[t] += red[t + off];
    __syncthreads();
  }
  const float inv = 1.f / red[0];
#pragma unroll
  for (int j = 0; j < 4; ++j) row[t + j * 256] = v[j] * inv;
}

// ys[b,p,c] = sum_n y1[b,c,p,n] * s[b,p,n]
__global__ __launch_bounds__(256)
void ys_k(const __bf16* __restrict__ y, const float* __restrict__ s,
          float* __restrict__ ys) {
  __shared__ float sL[NN];
  const int t = threadIdx.x;
  const int b = blockIdx.x >> 2, p = blockIdx.x & 3;
  const float* srow = s + (size_t)blockIdx.x * NN;
  for (int j = t; j < NN; j += 256) sL[j] = srow[j];
  __syncthreads();
  const __bf16* yb = y + (size_t)b * PER_BATCH;
  for (int c = t; c < NC; c += 256) {
    float acc = 0.f;
    for (int n = 0; n < NN; ++n) acc += bf2f(yb[act_idx(c, p * NN + n)]) * sL[n];
    ys[(size_t)blockIdx.x * NC + c] = acc;
  }
}

// cv[b,p,c] = W_k[c,:] . ys[b,p,:] + b_k[c]   (rows 1..512 of qkv_w)
__global__ __launch_bounds__(256)
void cv_k(const float* __restrict__ ys, const float* __restrict__ qkv_w,
          const float* __restrict__ qkv_b, float* __restrict__ cv) {
  __shared__ float yL[NC];
  const int t = threadIdx.x, r = blockIdx.x;
  yL[t] = ys[(size_t)r * NC + t];
  yL[t + 256] = ys[(size_t)r * NC + t + 256];
  __syncthreads();
  for (int c = t; c < NC; c += 256) {
    const float* wk = qkv_w + (size_t)(1 + c) * NC;
    float acc = qkv_b[1 + c];
    for (int cc = 0; cc < NC; ++cc) acc += wk[cc] * yL[cc];
    cv[(size_t)r * NC + c] = acc;
  }
}

// ---------------------------------------------------------------------------
// WMMA GEMM: Out[b, m, n] = epilogue( sum_k Wp[m,k] * Y[b,k,n] + bias[m] )
// Block tile 256(M) x 64(N): 8 waves, each owning two 16-row A bands sharing
// all four B fragments -> 8 WMMA per 12 ds_load_b128 per K-step.
// Double-buffered async staging: batch k+1 copies overlap tile-k WMMAs;
// each wave waits its own 5 in-flight ops (in-order) then barriers.
// MODE 0: relu(v)*cv -> bf16 swizzled   (v-proj + linear attention scaling)
// MODE 1: v + x      -> f32 natural     (out-proj + residual)
// MODE 2: silu(v)    -> bf16 swizzled   (ffn1)
// MODE 3: out += v   -> f32 natural     (ffn2 + residual accumulate)
// ---------------------------------------------------------------------------
template <int MODE>
__global__ __launch_bounds__(256)
void wmma_gemm_k(const __bf16* __restrict__ Wp, const __bf16* __restrict__ Y,
                 const float* __restrict__ bias, const float* __restrict__ extra,
                 void* __restrict__ Out, int M, int K) {
  __shared__ __align__(16) __bf16 lA[2 * 16 * 512];  // 32 KB double-buffered
  __shared__ __align__(16) __bf16 lB[2 * 4 * 512];   //  8 KB double-buffered
  const int t = threadIdx.x;
  const int b = blockIdx.z;
  const int mBase = blockIdx.y * 256;
  const int nBase = blockIdx.x * 64;
  const int mt0 = mBase >> 4;           // first A tile row index (16 tiles)
  const int nb  = nBase >> 6;           // B tile column index
  const int kTiles = K >> 5;
  const __bf16* Yb = Y + (size_t)b * (size_t)K * NPOS;
  const int wave = t >> 5, lane = t & 31;

  const unsigned ldsA = (unsigned)(uintptr_t)lA;
  const unsigned ldsB = (unsigned)(uintptr_t)lB;

  v8f zero = {0.f, 0.f, 0.f, 0.f, 0.f, 0.f, 0.f, 0.f};
  v8f acc[2][4] = {{zero, zero, zero, zero}, {zero, zero, zero, zero}};

  // Issue one K-step's staging (5 async instructions per wave).
  auto stage = [&](int kb, int sel) {
#pragma unroll
    for (int u = 0; u < 4; ++u) {               // A: 16 tiles x 1KB
      const int i = t + (u << 8);               // 0..1023
      const int ch = i >> 6;                    // tile 0..15
      const int off = (i & 63) << 4;            // byte offset in tile
      const char* src =
          (const char*)(Wp + (((size_t)(mt0 + ch) * kTiles + kb) << 9)) + off;
      async_copy16(ldsA + (unsigned)((sel << 14) + (ch << 10) + off), src);
    }
    // B: one contiguous 4KB tile
    const char* srcB =
        (const char*)(Yb + ((size_t)(kb * 64 + nb) << 11)) + (t << 4);
    async_copy16(ldsB + (unsigned)((sel << 12) + (t << 4)), srcB);
  };

  stage(0, 0);
  for (int kb = 0; kb < kTiles; ++kb) {
    const int sel = kb & 1;
    if (kb + 1 < kTiles) {
      stage(kb + 1, sel ^ 1);
      asm volatile("s_wait_asynccnt 0x5" ::: "memory");  // batch kb landed
    } else {
      asm volatile("s_wait_asynccnt 0x0" ::: "memory");
    }
    __syncthreads();

    // Fragments: each lane's 16 bf16 are contiguous -> 2x ds_load_b128 each.
    v16bf a[2];
#pragma unroll
    for (int h = 0; h < 2; ++h) {
      const __bf16* aP = lA + (sel << 13) + ((wave + (h << 3)) << 9) + (lane << 4);
      v8bf a0 = *(const v8bf*)aP;
      v8bf a1 = *(const v8bf*)(aP + 8);
      a[h] = __builtin_shufflevector(a0, a1, 0, 1, 2, 3, 4, 5, 6, 7,
                                     8, 9, 10, 11, 12, 13, 14, 15);
    }
#pragma unroll
    for (int nt = 0; nt < 4; ++nt) {
      const __bf16* bP = lB + (sel << 11) + (nt << 9) + (lane << 4);
      v8bf b0 = *(const v8bf*)bP;
      v8bf b1 = *(const v8bf*)(bP + 8);
      v16bf bb = __builtin_shufflevector(b0, b1, 0, 1, 2, 3, 4, 5, 6, 7,
                                         8, 9, 10, 11, 12, 13, 14, 15);
      acc[0][nt] = __builtin_amdgcn_wmma_f32_16x16x32_bf16(
          false, a[0], false, bb, (short)0, acc[0][nt], false, false);
      acc[1][nt] = __builtin_amdgcn_wmma_f32_16x16x32_bf16(
          false, a[1], false, bb, (short)0, acc[1][nt], false, false);
    }
    __syncthreads();   // protect buffer reuse two iterations later
  }

  // Epilogue.  D layout: VGPR r -> M = r (lanes 0-15) / r+8 (lanes 16-31).
  const int no = nBase + (lane & 15);
  const size_t bOff = (size_t)b * (size_t)M * NPOS;
#pragma unroll
  for (int h = 0; h < 2; ++h) {
    const int mo = mBase + (wave + (h << 3)) * 16 + ((lane >= 16) ? 8 : 0);
#pragma unroll
    for (int nt = 0; nt < 4; ++nt) {
      const int n = no + nt * 16;
      if (MODE == 0 || MODE == 2) {
        v8bf o;
#pragma unroll
        for (int r = 0; r < 8; ++r) {
          float v = acc[h][nt][r] + bias[mo + r];
          if (MODE == 0)
            v = fmaxf(v, 0.f) * extra[((size_t)b * NP + (n >> 10)) * NC + mo + r];
          else
            v = v / (1.f + __expf(-v));
          o[r] = f2bf(v);
        }
        // 8 consecutive j-slots in the swizzled layout -> one 16B store
        *(v8bf*)(((__bf16*)Out) + bOff + act_idx(mo, n)) = o;
      } else {
#pragma unroll
        for (int r = 0; r < 8; ++r) {
          const int m = mo + r;
          float v = acc[h][nt][r] + bias[m];
          const size_t oi = bOff + (size_t)m * NPOS + n;
          if (MODE == 1) {
            ((float*)Out)[oi] = v + extra[oi];
          } else {
            ((float*)Out)[oi] += v;
          }
        }
      }
    }
  }
}

// ---------------------------------------------------------------------------
// Host-side orchestration.
// Workspace layout:
//   [0, 1MB)        : partials(2048f) stats1(32f) stats2(32f) q(65536f)
//                     ys(32768f) cv(32768f)
//   [1MB, 4MB)      : packed bf16 weights (wv, wo, wf1, wf2)
//   [4MB, 68MB)     : y1 bf16 swizzled
//   [68MB, 132MB)   : g  bf16 swizzled (relu(v)*cv), reused as y2
//   [132MB, 260MB)  : h  bf16 swizzled
// ---------------------------------------------------------------------------
extern "C" void kernel_launch(void* const* d_in, const int* in_sizes, int n_in,
                              void* d_out, int out_size, void* d_ws, size_t ws_size,
                              hipStream_t stream) {
  (void)in_sizes; (void)n_in; (void)out_size; (void)ws_size;
  const float* x     = (const float*)d_in[0];
  const float* gn1_g = (const float*)d_in[1];
  const float* gn1_b = (const float*)d_in[2];
  const float* qkv_w = (const float*)d_in[3];
  const float* qkv_b = (const float*)d_in[4];
  const float* out_w = (const float*)d_in[5];
  const float* out_b = (const float*)d_in[6];
  const float* gn2_g = (const float*)d_in[7];
  const float* gn2_b = (const float*)d_in[8];
  const float* f1_w  = (const float*)d_in[9];
  const float* f1_b  = (const float*)d_in[10];
  const float* f2_w  = (const float*)d_in[11];
  const float* f2_b  = (const float*)d_in[12];
  float* out = (float*)d_out;

  float* wsF      = (float*)d_ws;
  float* partials = wsF;               // 2048 floats
  float* stats1   = wsF + 2048;
  float* stats2   = wsF + 2080;
  float* qbuf     = wsF + 4096;        // 65536
  float* ysbuf    = qbuf + 65536;      // 32768
  float* cvbuf    = ysbuf + 32768;     // 32768

  __bf16* wv_p  = (__bf16*)((char*)d_ws + ((size_t)1 << 20));
  __bf16* wo_p  = wv_p + 512 * 512;
  __bf16* wf1_p = wo_p + 512 * 512;
  __bf16* wf2_p = wf1_p + 1024 * 512;

  __bf16* y1   = (__bf16*)((char*)d_ws + ((size_t)4 << 20));
  __bf16* gbuf = y1 + (size_t)33554432;      // reused as y2
  __bf16* hbuf = gbuf + (size_t)33554432;

  // --- weight repack (one-time per call, tiny) ---
  wpack_k<<<1024, 256, 0, stream>>>(qkv_w + (size_t)(1 + NC) * NC, wv_p, 9);
  wpack_k<<<1024, 256, 0, stream>>>(out_w, wo_p, 9);
  wpack_k<<<2048, 256, 0, stream>>>(f1_w, wf1_p, 9);
  wpack_k<<<2048, 256, 0, stream>>>(f2_w, wf2_p, 10);

  // --- GroupNorm 1 -> y1 (swizzled bf16) ---
  gn_partial_k<<<NB * 64, 256, 0, stream>>>(x, partials);
  gn_finalize_k<<<NB, 64, 0, stream>>>(partials, stats1);
  norm_pack_k<<<16384, 256, 0, stream>>>(x, stats1, gn1_g, gn1_b, y1);

  // --- linear attention scalars: q GEMV, softmax, weighted reduce, cv ---
  q_k<<<dim3(NPOS / 256, NB), 256, 0, stream>>>(y1, qkv_w, qkv_b, qbuf);
  softmax_k<<<NB * NP, 256, 0, stream>>>(qbuf);
  ys_k<<<NB * NP, 256, 0, stream>>>(y1, qbuf, ysbuf);
  cv_k<<<NB * NP, 256, 0, stream>>>(ysbuf, qkv_w, qkv_b, cvbuf);

  // --- v-proj GEMM + relu*cv -> g ---
  wmma_gemm_k<0><<<dim3(NPOS / 64, NC / 256, NB), 256, 0, stream>>>(
      wv_p, y1, qkv_b + 1 + NC, cvbuf, gbuf, NC, NC);

  // --- out-proj GEMM + residual -> d_out ---
  wmma_gemm_k<1><<<dim3(NPOS / 64, NC / 256, NB), 256, 0, stream>>>(
      wo_p, gbuf, out_b, x, out, NC, NC);

  // --- GroupNorm 2 (on x2 = d_out) -> y2 (reuse g region) ---
  gn_partial_k<<<NB * 64, 256, 0, stream>>>(out, partials);
  gn_finalize_k<<<NB, 64, 0, stream>>>(partials, stats2);
  norm_pack_k<<<16384, 256, 0, stream>>>(out, stats2, gn2_g, gn2_b, gbuf);

  // --- ffn1 GEMM + SiLU -> h ---
  wmma_gemm_k<2><<<dim3(NPOS / 64, NF / 256, NB), 256, 0, stream>>>(
      wf1_p, gbuf, f1_b, nullptr, hbuf, NF, NC);

  // --- ffn2 GEMM + accumulate into d_out ---
  wmma_gemm_k<3><<<dim3(NPOS / 64, NC / 256, NB), 256, 0, stream>>>(
      wf2_p, hbuf, f2_b, nullptr, out, NC, NF);
}
